// NuGraphCore_74148315398249
// MI455X (gfx1250) — matrix operand order repaired and checked by reading, not compiled, not run on hardware
//
#include <hip/hip_runtime.h>
#include <math.h>

// ---------------------------------------------------------------------------
// NuGraphCore forward for MI455X (gfx1250): wave32, WMMA f16->f32 GEMMs,
// float atomics for segment-softmax aggregation, fast HW transcendentals.
// ---------------------------------------------------------------------------

typedef _Float16 half_t;
typedef __attribute__((ext_vector_type(16))) _Float16 v16h;
typedef __attribute__((ext_vector_type(8)))  _Float16 v8h;
typedef __attribute__((ext_vector_type(8)))  float    v8f;

#define N_HIT   200000
#define N_SP    40000
#define N_EVT   512
#define NE_PLAN 1200000
#define NE_NEX  400000
#define NE_SPE  40000

// ---------------------------------------------------------------------------
// device helpers (hardware v_exp_f32 / v_rcp_f32 instead of libm chains)
// ---------------------------------------------------------------------------
__device__ __forceinline__ float mishf(float x) {
    // mish(x) = x * tanh(softplus(x));  e^{2*softplus(x)} = (1+e^x)^2
    // tanh(softplus(x)) = (w-1)/(w+1), w = (1+e^x)^2.
    if (x > 20.f) return x;              // factor == 1 to fp32 precision
    float u = __expf(x);                 // v_exp_f32
    float w = 1.f + u;
    w = w * w;
    return x * (w - 1.f) * __builtin_amdgcn_rcpf(w + 1.f);
}
__device__ __forceinline__ float sigmoidf(float x) {
    return __builtin_amdgcn_rcpf(1.f + __expf(-x));
}

__device__ __forceinline__ v8f wmma16(v16h a, v16h b, v8f c) {
    // D = A(16x32 f16) * B(32x16 f16) + C(16x16 f32)
    return __builtin_amdgcn_wmma_f32_16x16x32_f16(
        /*neg_a=*/false, a, /*neg_b=*/false, b,
        /*c_mod=*/(short)0, c, /*reuse_a=*/false, /*reuse_b=*/false);
}

// A fragment (16x32 slice, row-major f16 source [16][ldk]).
// ISA 16-bit A layout: lanes 0-15 row M=lane, halves 0-7 -> K=kb..kb+7,
// halves 8-15 -> K=kb+16..kb+23; lanes 16-31 same rows, K offset +8 / +24.
__device__ __forceinline__ v16h load_a_frag(const half_t* base, int ldk, int kstep) {
    int lane = threadIdx.x & 31;
    int m    = lane & 15;
    int kb   = kstep * 32 + (lane >> 4) * 8;
    const half_t* p = base + m * ldk + kb;
    union { v16h v; v8h h[2]; } u;
    u.h[0] = *(const v8h*)(p);
    u.h[1] = *(const v8h*)(p + 16);
    return u.v;
}

// B fragment (32x16 slice) from K-major repacked weights Wt[N][ldk] (f16):
// lanes 0-15 column N = n0+lane, K = kb..kb+7, kb+16..kb+23; lanes 16-31 K offset +8 / +24.
__device__ __forceinline__ v16h load_b_frag(const half_t* wt, int n0, int ldk, int kstep) {
    int lane = threadIdx.x & 31;
    int n    = n0 + (lane & 15);
    int kb   = kstep * 32 + (lane >> 4) * 8;
    const half_t* p = wt + n * ldk + kb;
    union { v16h v; v8h h[2]; } u;
    u.h[0] = *(const v8h*)(p);
    u.h[1] = *(const v8h*)(p + 16);
    return u.v;
}

// ---------------------------------------------------------------------------
// small utility kernels
// ---------------------------------------------------------------------------
__global__ void fill_kernel(float* p, int n, float v) {
    int i = blockIdx.x * 256 + threadIdx.x;
    if (i < n) p[i] = v;
}

// repack f32 [K][N] -> f16 [N][Kpad] (zero padded rows K..Kpad)
__global__ void wtrans_kernel(half_t* dst, const float* src, int K, int N, int Kpad) {
    int i = blockIdx.x * 256 + threadIdx.x;
    if (i >= N * Kpad) return;
    int n = i / Kpad, k = i - n * Kpad;
    dst[i] = (k < K) ? (half_t)src[k * N + n] : (half_t)0.f;
}

// ---------------------------------------------------------------------------
// edge gate: gate[e] = sigmoid([x_dst[dst], x_src[src]] . We + be)
// one wave per edge, 2 feats/lane, shfl reduction (wave32)
// ---------------------------------------------------------------------------
__global__ void gate_kernel(const float* __restrict__ xsrc, const float* __restrict__ xdst,
                            const int* __restrict__ esrc, const int* __restrict__ edst,
                            const float* __restrict__ We, const float* __restrict__ be,
                            float* __restrict__ gate, int E) {
    int wid  = (blockIdx.x * blockDim.x + threadIdx.x) >> 5;
    int lane = threadIdx.x & 31;
    if (wid >= E) return;
    int s = esrc[wid], d = edst[wid];
    float2 wi = *(const float2*)(We + 2 * lane);        // rows 0..63  (dst feats)
    float2 wj = *(const float2*)(We + 64 + 2 * lane);   // rows 64..127 (src feats)
    float2 xi = *(const float2*)(xdst + (long long)d * 64 + 2 * lane);
    float2 xj = *(const float2*)(xsrc + (long long)s * 64 + 2 * lane);
    float p = xi.x * wi.x + xi.y * wi.y + xj.x * wj.x + xj.y * wj.y;
    for (int off = 16; off > 0; off >>= 1) p += __shfl_down(p, off, 32);
    if (lane == 0) gate[wid] = sigmoidf(p + be[0]);
}

// ---------------------------------------------------------------------------
// segment-softmax scatter passes (thread = one (edge, feature) pair)
// ---------------------------------------------------------------------------
__global__ void scatter_max_kernel(const float* __restrict__ xsrc,
                                   const int* __restrict__ esrc, const int* __restrict__ edst,
                                   const float* __restrict__ gate,
                                   float* __restrict__ mx, int total) {
    int i = blockIdx.x * 256 + threadIdx.x;
    if (i >= total) return;
    int e = i >> 6, f = i & 63;
    float m = gate[e] * xsrc[(long long)esrc[e] * 64 + f];
    atomicMax(&mx[(long long)edst[e] * 64 + f], m);   // global_atomic_max_num_f32
}

__global__ void scatter_expsum_kernel(const float* __restrict__ xsrc,
                                      const int* __restrict__ esrc, const int* __restrict__ edst,
                                      const float* __restrict__ gate,
                                      const float* __restrict__ mx,
                                      float* __restrict__ denom, int total) {
    int i = blockIdx.x * 256 + threadIdx.x;
    if (i >= total) return;
    int e = i >> 6, f = i & 63;
    long long di = (long long)edst[e] * 64 + f;
    float m = gate[e] * xsrc[(long long)esrc[e] * 64 + f];
    atomicAdd(&denom[di], __expf(m - mx[di]));
}

__global__ void scatter_wsum_kernel(const float* __restrict__ xsrc,
                                    const int* __restrict__ esrc, const int* __restrict__ edst,
                                    const float* __restrict__ gate,
                                    const float* __restrict__ mx,
                                    const float* __restrict__ denom,
                                    float* __restrict__ aggr, int total) {
    int i = blockIdx.x * 256 + threadIdx.x;
    if (i >= total) return;
    int e = i >> 6, f = i & 63;
    long long di = (long long)edst[e] * 64 + f;
    float m = gate[e] * xsrc[(long long)esrc[e] * 64 + f];
    float w = __expf(m - mx[di]) * __builtin_amdgcn_rcpf(denom[di]);
    atomicAdd(&aggr[di], m * w);
}

// ---------------------------------------------------------------------------
// node MLP of a block: out = mish(mish([aggr, xdst] @ W1 + b1) @ W2 + b2)
// 16-row tile per workgroup, 4 waves each owning 16 output columns.
// ---------------------------------------------------------------------------
__global__ void block_mlp_kernel(const float* __restrict__ aggr, const float* __restrict__ xdst,
                                 const half_t* __restrict__ w1t,   // [64][128] f16, K-major
                                 const float* __restrict__ b1,
                                 const half_t* __restrict__ w2t,   // [64][64]  f16, K-major
                                 const float* __restrict__ b2,
                                 float* __restrict__ out) {
    __shared__ __align__(16) half_t As[16 * 128];
    __shared__ __align__(16) half_t Hs[16 * 64];
    int row0 = blockIdx.x * 16;
    int t = threadIdx.x;                               // 128 threads
    for (int i = t; i < 16 * 128; i += 128) {
        int r = i >> 7, c = i & 127;
        float v = (c < 64) ? aggr[(long long)(row0 + r) * 64 + c]
                           : xdst[(long long)(row0 + r) * 64 + (c - 64)];
        As[i] = (half_t)v;
    }
    __syncthreads();
    int wave = t >> 5, lane = t & 31;
    int n0 = wave * 16;
    int col = n0 + (lane & 15);
    int mofs = (lane >> 4) << 3;

    v8f acc = {};
    for (int ks = 0; ks < 4; ++ks)
        acc = wmma16(load_a_frag(As, 128, ks), load_b_frag(w1t, n0, 128, ks), acc);
    float bias = b1[col];
    for (int r = 0; r < 8; ++r)
        Hs[(r + mofs) * 64 + col] = (half_t)mishf(acc[r] + bias);
    __syncthreads();

    v8f acc2 = {};
    for (int ks = 0; ks < 2; ++ks)
        acc2 = wmma16(load_a_frag(Hs, 64, ks), load_b_frag(w2t, n0, 64, ks), acc2);
    float bias2 = b2[col];
    for (int r = 0; r < 8; ++r)
        out[(long long)(row0 + r + mofs) * 64 + col] = mishf(acc2[r] + bias2);
}

// ---------------------------------------------------------------------------
// beta head: sigmoid(mlp3([of, h]))   (65 -> 192 -> 192 -> 1)
// ---------------------------------------------------------------------------
__global__ void mlp3_beta_kernel(const float* __restrict__ of, const float* __restrict__ h,
                                 const half_t* __restrict__ w1t,  // [192][96]
                                 const float* __restrict__ b1,
                                 const half_t* __restrict__ w2t,  // [192][192]
                                 const float* __restrict__ b2,
                                 const float* __restrict__ W3,    // [192] f32
                                 const float* __restrict__ b3,
                                 float* __restrict__ out) {
    __shared__ __align__(16) half_t As[16 * 96];
    __shared__ __align__(16) half_t H1[16 * 192];
    __shared__ __align__(16) half_t H2[16 * 192];
    int row0 = blockIdx.x * 16;
    int t = threadIdx.x;                               // 128 threads
    for (int i = t; i < 16 * 96; i += 128) {
        int r = i / 96, c = i - r * 96;
        float v = 0.f;
        if (c == 0)       v = of[row0 + r];
        else if (c < 65)  v = h[(long long)(row0 + r) * 64 + (c - 1)];
        As[i] = (half_t)v;
    }
    __syncthreads();
    int wave = t >> 5, lane = t & 31;
    int mofs = (lane >> 4) << 3;

    for (int nt = 0; nt < 3; ++nt) {
        int n0 = (wave * 3 + nt) * 16;
        v8f acc = {};
        for (int ks = 0; ks < 3; ++ks)
            acc = wmma16(load_a_frag(As, 96, ks), load_b_frag(w1t, n0, 96, ks), acc);
        int col = n0 + (lane & 15);
        float bias = b1[col];
        for (int r = 0; r < 8; ++r)
            H1[(r + mofs) * 192 + col] = (half_t)mishf(acc[r] + bias);
    }
    __syncthreads();
    for (int nt = 0; nt < 3; ++nt) {
        int n0 = (wave * 3 + nt) * 16;
        v8f acc = {};
        for (int ks = 0; ks < 6; ++ks)
            acc = wmma16(load_a_frag(H1, 192, ks), load_b_frag(w2t, n0, 192, ks), acc);
        int col = n0 + (lane & 15);
        float bias = b2[col];
        for (int r = 0; r < 8; ++r)
            H2[(r + mofs) * 192 + col] = (half_t)mishf(acc[r] + bias);
    }
    __syncthreads();
    if (t < 16) {
        float s = b3[0];
        for (int k = 0; k < 192; ++k) s += (float)H2[t * 192 + k] * W3[k];
        out[row0 + t] = sigmoidf(s);
    }
}

// ---------------------------------------------------------------------------
// coord head: mlp3([ox, h])   (80 -> 192 -> 192 -> 16), no sigmoid
// ---------------------------------------------------------------------------
__global__ void mlp3_coord_kernel(const float* __restrict__ ox, const float* __restrict__ h,
                                  const half_t* __restrict__ w1t,  // [192][96]
                                  const float* __restrict__ b1,
                                  const half_t* __restrict__ w2t,  // [192][192]
                                  const float* __restrict__ b2,
                                  const half_t* __restrict__ w3t,  // [16][192]
                                  const float* __restrict__ b3,
                                  float* __restrict__ out) {
    __shared__ __align__(16) half_t As[16 * 96];
    __shared__ __align__(16) half_t H1[16 * 192];
    __shared__ __align__(16) half_t H2[16 * 192];
    int row0 = blockIdx.x * 16;
    int t = threadIdx.x;
    for (int i = t; i < 16 * 96; i += 128) {
        int r = i / 96, c = i - r * 96;
        float v = 0.f;
        if (c < 16)       v = ox[(long long)(row0 + r) * 16 + c];
        else if (c < 80)  v = h[(long long)(row0 + r) * 64 + (c - 16)];
        As[i] = (half_t)v;
    }
    __syncthreads();
    int wave = t >> 5, lane = t & 31;
    int mofs = (lane >> 4) << 3;

    for (int nt = 0; nt < 3; ++nt) {
        int n0 = (wave * 3 + nt) * 16;
        v8f acc = {};
        for (int ks = 0; ks < 3; ++ks)
            acc = wmma16(load_a_frag(As, 96, ks), load_b_frag(w1t, n0, 96, ks), acc);
        int col = n0 + (lane & 15);
        float bias = b1[col];
        for (int r = 0; r < 8; ++r)
            H1[(r + mofs) * 192 + col] = (half_t)mishf(acc[r] + bias);
    }
    __syncthreads();
    for (int nt = 0; nt < 3; ++nt) {
        int n0 = (wave * 3 + nt) * 16;
        v8f acc = {};
        for (int ks = 0; ks < 6; ++ks)
            acc = wmma16(load_a_frag(H1, 192, ks), load_b_frag(w2t, n0, 192, ks), acc);
        int col = n0 + (lane & 15);
        float bias = b2[col];
        for (int r = 0; r < 8; ++r)
            H2[(r + mofs) * 192 + col] = (half_t)mishf(acc[r] + bias);
    }
    __syncthreads();
    if (wave == 0) {
        v8f acc = {};
        for (int ks = 0; ks < 6; ++ks)
            acc = wmma16(load_a_frag(H2, 192, ks), load_b_frag(w3t, 0, 192, ks), acc);
        int col = lane & 15;
        float bias = b3[col];
        for (int r = 0; r < 8; ++r)
            out[(long long)(row0 + r + mofs) * 16 + col] = acc[r] + bias;
    }
}

// ---------------------------------------------------------------------------
// host orchestration
// ---------------------------------------------------------------------------
struct BlockW { const float *We, *be, *W1, *b1, *W2, *b2; };

static void run_block(hipStream_t s,
                      const float* xsrc, const float* xdst,
                      const int* esrc, const int* edst, int E, int Ndst,
                      const BlockW& w, const half_t* w1t, const half_t* w2t,
                      float* gate, float* mx, float* denom, float* aggr,
                      float* out) {
    int nf = Ndst * 64;
    int ef = E * 64;
    gate_kernel<<<(E * 32 + 255) / 256, 256, 0, s>>>(xsrc, xdst, esrc, edst, w.We, w.be, gate, E);
    fill_kernel<<<(nf + 255) / 256, 256, 0, s>>>(mx, nf, -INFINITY);
    fill_kernel<<<(nf + 255) / 256, 256, 0, s>>>(denom, nf, 0.f);
    fill_kernel<<<(nf + 255) / 256, 256, 0, s>>>(aggr, nf, 0.f);
    scatter_max_kernel<<<(ef + 255) / 256, 256, 0, s>>>(xsrc, esrc, edst, gate, mx, ef);
    scatter_expsum_kernel<<<(ef + 255) / 256, 256, 0, s>>>(xsrc, esrc, edst, gate, mx, denom, ef);
    scatter_wsum_kernel<<<(ef + 255) / 256, 256, 0, s>>>(xsrc, esrc, edst, gate, mx, denom, aggr, ef);
    block_mlp_kernel<<<Ndst / 16, 128, 0, s>>>(aggr, xdst, w1t, w.b1, w2t, w.b2, out);
}

extern "C" void kernel_launch(void* const* d_in, const int* in_sizes, int n_in,
                              void* d_out, int out_size, void* d_ws, size_t ws_size,
                              hipStream_t stream) {
    (void)in_sizes; (void)n_in; (void)out_size; (void)ws_size;

    const float* x_hit = (const float*)d_in[0];
    const float* x_sp  = (const float*)d_in[1];
    const float* x_evt = (const float*)d_in[2];
    const float* of    = (const float*)d_in[3];
    const float* ox    = (const float*)d_in[4];

    BlockW blk[5];  // plane, p2n, n2i, i2n, n2p
    for (int i = 0; i < 5; ++i) {
        int b = 5 + 6 * i;
        blk[i].We = (const float*)d_in[b + 0];
        blk[i].be = (const float*)d_in[b + 1];
        blk[i].W1 = (const float*)d_in[b + 2];
        blk[i].b1 = (const float*)d_in[b + 3];
        blk[i].W2 = (const float*)d_in[b + 4];
        blk[i].b2 = (const float*)d_in[b + 5];
    }
    const float* beta_W1 = (const float*)d_in[35];
    const float* beta_b1 = (const float*)d_in[36];
    const float* beta_W2 = (const float*)d_in[37];
    const float* beta_b2 = (const float*)d_in[38];
    const float* beta_W3 = (const float*)d_in[39];
    const float* beta_b3 = (const float*)d_in[40];
    const float* crd_W1  = (const float*)d_in[41];
    const float* crd_b1  = (const float*)d_in[42];
    const float* crd_W2  = (const float*)d_in[43];
    const float* crd_b2  = (const float*)d_in[44];
    const float* crd_W3  = (const float*)d_in[45];
    const float* crd_b3  = (const float*)d_in[46];
    const int* e_plan = (const int*)d_in[47];
    const int* e_nex  = (const int*)d_in[48];
    const int* e_spe  = (const int*)d_in[49];

    // output regions (tuple concat: h, sp, evt, of_new, ox_new)
    float* out_h   = (float*)d_out;
    float* out_sp  = out_h  + (size_t)N_HIT * 64;
    float* out_evt = out_sp + (size_t)N_SP  * 64;
    float* out_of  = out_evt + (size_t)N_EVT * 64;
    float* out_ox  = out_of + (size_t)N_HIT;

    // workspace carve-up (256B aligned)
    char* wp = (char*)d_ws;
    auto carve = [&](size_t bytes) -> void* {
        void* p = (void*)wp;
        wp += (bytes + 255) & ~(size_t)255;
        return p;
    };
    float* gate  = (float*)carve((size_t)NE_PLAN * 4);
    float* mx    = (float*)carve((size_t)N_HIT * 64 * 4);
    float* denom = (float*)carve((size_t)N_HIT * 64 * 4);
    float* aggr  = (float*)carve((size_t)N_HIT * 64 * 4);
    half_t* w1t[5]; half_t* w2t[5];
    for (int i = 0; i < 5; ++i) {
        w1t[i] = (half_t*)carve((size_t)64 * 128 * 2);
        w2t[i] = (half_t*)carve((size_t)64 * 64 * 2);
    }
    half_t* beta_w1t = (half_t*)carve((size_t)192 * 96 * 2);
    half_t* beta_w2t = (half_t*)carve((size_t)192 * 192 * 2);
    half_t* crd_w1t  = (half_t*)carve((size_t)192 * 96 * 2);
    half_t* crd_w2t  = (half_t*)carve((size_t)192 * 192 * 2);
    half_t* crd_w3t  = (half_t*)carve((size_t)16 * 192 * 2);

    // repack weights: f32 [K][N] -> f16 [N][Kpad]
    auto wt = [&](half_t* dst, const float* src, int K, int N, int Kpad) {
        int n = N * Kpad;
        wtrans_kernel<<<(n + 255) / 256, 256, 0, stream>>>(dst, src, K, N, Kpad);
    };
    for (int i = 0; i < 5; ++i) {
        wt(w1t[i], blk[i].W1, 128, 64, 128);
        wt(w2t[i], blk[i].W2, 64, 64, 64);
    }
    wt(beta_w1t, beta_W1, 65, 192, 96);
    wt(beta_w2t, beta_W2, 192, 192, 192);
    wt(crd_w1t,  crd_W1,  80, 192, 96);
    wt(crd_w2t,  crd_W2,  192, 192, 192);
    wt(crd_w3t,  crd_W3,  192, 16, 192);

    // edge rows: [0]=source, [1]=target
    const int* plan_s = e_plan;            const int* plan_d = e_plan + NE_PLAN;
    const int* nex_s  = e_nex;             const int* nex_d  = e_nex + NE_NEX;
    const int* spe_s  = e_spe;             const int* spe_d  = e_spe + NE_SPE;

    // 1) plane:  hit->hit
    run_block(stream, x_hit, x_hit, plan_s, plan_d, NE_PLAN, N_HIT,
              blk[0], w1t[0], w2t[0], gate, mx, denom, aggr, out_h);
    // 2) p2n: h -> sp
    run_block(stream, out_h, x_sp, nex_s, nex_d, NE_NEX, N_SP,
              blk[1], w1t[1], w2t[1], gate, mx, denom, aggr, out_sp);
    // 3) n2i: sp -> evt
    run_block(stream, out_sp, x_evt, spe_s, spe_d, NE_SPE, N_EVT,
              blk[2], w1t[2], w2t[2], gate, mx, denom, aggr, out_evt);
    // 4) i2n: evt -> sp (reversed edges), in-place on out_sp
    run_block(stream, out_evt, out_sp, spe_d, spe_s, NE_SPE, N_SP,
              blk[3], w1t[3], w2t[3], gate, mx, denom, aggr, out_sp);
    // 5) n2p: sp -> hit (reversed edges), in-place on out_h
    run_block(stream, out_sp, out_h, nex_d, nex_s, NE_NEX, N_HIT,
              blk[4], w1t[4], w2t[4], gate, mx, denom, aggr, out_h);

    // heads
    mlp3_beta_kernel<<<N_HIT / 16, 128, 0, stream>>>(
        of, out_h, beta_w1t, beta_b1, beta_w2t, beta_b2, beta_W3, beta_b3, out_of);
    mlp3_coord_kernel<<<N_HIT / 16, 128, 0, stream>>>(
        ox, out_h, crd_w1t, crd_b1, crd_w2t, crd_b2, crd_w3t, crd_b3, out_ox);
}
